// BDH_GPU_90692529422839
// MI455X (gfx1250) — compile-verified, matrix operand order checked by reading, not run on hardware
//
#include <hip/hip_runtime.h>
#include <hip/hip_bf16.h>

// ---------------- problem constants (match reference) ----------------
constexpr int Dm   = 1024;   // D
constexpr int Hh   = 16;     // H
constexpr int Nf   = 8192;   // N
constexpr int VOC  = 32000;  // VOCAB
constexpr int Lit  = 4;      // L
constexpr int SDm  = 512;    // SD = N/H
constexpr int Bb   = 2;
constexpr int Tt   = 1024;
constexpr int BT   = Bb * Tt;   // 2048

typedef unsigned short u16;
typedef __attribute__((ext_vector_type(16))) __bf16 v16bf;
typedef __attribute__((ext_vector_type(8)))  float  v8f;

union FragU { v16bf v; uint4 q[2]; };
union AccU  { v8f   v; float f[8]; };

__device__ __forceinline__ u16 f2bf(float f) {
  unsigned u  = __float_as_uint(f);
  unsigned rb = ((u >> 16) & 1u) + 0x7fffu;   // round-to-nearest-even
  return (u16)((u + rb) >> 16);
}
__device__ __forceinline__ float bf2f(u16 h) {
  return __uint_as_float(((unsigned)h) << 16);
}

// LDS byte offset of a __shared__-derived pointer: LDS aperture keeps the
// segment offset in the low 32 bits of the generic address.
__device__ __forceinline__ unsigned lds_off(const void* p) {
  return (unsigned)(unsigned long long)p;
}

// CDNA5 async copy: memory -> LDS, 16B per lane, tracked by ASYNCcnt.
__device__ __forceinline__ void async_ld_b128(unsigned lds_byte_off, const void* gaddr) {
  asm volatile("global_load_async_to_lds_b128 %0, %1, off"
               :: "v"(lds_byte_off), "v"(gaddr)
               : "memory");
}
__device__ __forceinline__ void wait_async0() {
  asm volatile("s_wait_asynccnt 0x0" ::: "memory");
}

// ---------------- generic bf16 WMMA strided-batched GEMM ----------------
// C[M,N] = op( A[M,K] row-major  x  Bt[N,K] row-major (i.e. B^T) ) per batch z.
// batch offsets: off = (z/bdiv)*s?0 + (z%bdiv)*s?1   (elements)
// Block tile 128x128xK32, 8 waves; each wave computes 32x64 via 2x4 WMMA accs.
#define GF_RELU    1
#define GF_CAUSAL  2   // keep col < row (tril k=-1), else 0
#define GF_OUTBF16 4
#define GF_AUX     8   // multiply elementwise by bf16 aux (post-ReLU)

constexpr int BM = 128, BN = 128, BK = 32;
constexpr int LS = BK + 8;   // LDS row stride (elements; 80B = 5x16B, conflict-free-ish)

__device__ __forceinline__ v16bf lds_frag(const u16* base, int row, int kbase) {
  FragU f;
  const u16* p = base + row * LS + kbase;
  f.q[0] = *(const uint4*)(p);
  f.q[1] = *(const uint4*)(p + 16);
  return f.v;
}

__global__ __launch_bounds__(256)
void gemm_bf16_wmma(const u16* __restrict__ A, const u16* __restrict__ Btr,
                    void* __restrict__ C, const u16* __restrict__ aux,
                    int M, int Nn, int K,
                    int lda, int ldb, int ldc, int ldaux,
                    int bdiv,
                    long long sA0, long long sA1,
                    long long sB0, long long sB1,
                    long long sC0, long long sC1,
                    long long sX0, long long sX1,
                    int flags)
{
  __shared__ __align__(16) u16 lAt[2][BM * LS];
  __shared__ __align__(16) u16 lBt[2][BN * LS];

  const int tid  = threadIdx.x;
  const int lane = tid & 31;
  const int wave = tid >> 5;
  const int wm   = wave & 3;       // 4 waves along M (32 rows each)
  const int wn   = wave >> 2;      // 2 waves along N (64 cols each)
  const int hl   = lane >> 4;      // half-wave select
  const int lN   = lane & 15;

  const int z = blockIdx.z;
  const int q = z / bdiv, r = z % bdiv;
  A   += (long long)q * sA0 + (long long)r * sA1;
  Btr += (long long)q * sB0 + (long long)r * sB1;
  const long long coff = (long long)q * sC0 + (long long)r * sC1;
  const long long xoff = (long long)q * sX0 + (long long)r * sX1;

  const int m0 = blockIdx.y * BM;
  const int n0 = blockIdx.x * BN;

  // per-thread staging coordinates (16B chunks): 128 rows x 32 cols per operand
  const int rr0 = tid >> 2;        // 0..63
  const int c8  = (tid & 3) * 8;   // 0..24

  const unsigned aBase0 = lds_off(&lAt[0][0]);
  const unsigned aBase1 = lds_off(&lAt[1][0]);
  const unsigned bBase0 = lds_off(&lBt[0][0]);
  const unsigned bBase1 = lds_off(&lBt[1][0]);

  auto issue_tile = [&](int k0, int buf) {
    const unsigned aB = buf ? aBase1 : aBase0;
    const unsigned bB = buf ? bBase1 : bBase0;
    #pragma unroll
    for (int p = 0; p < 2; ++p) {
      const int rr = rr0 + p * 64;
      async_ld_b128(aB + (unsigned)(rr * LS + c8) * 2,
                    A + (long long)(m0 + rr) * lda + (k0 + c8));
    }
    #pragma unroll
    for (int p = 0; p < 2; ++p) {
      const int rr = rr0 + p * 64;
      async_ld_b128(bB + (unsigned)(rr * LS + c8) * 2,
                    Btr + (long long)(n0 + rr) * ldb + (k0 + c8));
    }
  };

  v8f acc[2][4] = {};

  const int nk = K / BK;
  issue_tile(0, 0);

  for (int kt = 0; kt < nk; ++kt) {
    wait_async0();        // this wave's copies for tile kt are in LDS
    __syncthreads();      // whole tile resident; everyone done reading kt-1
    if (kt + 1 < nk) issue_tile((kt + 1) * BK, (kt + 1) & 1);

    const u16* lA = (kt & 1) ? &lAt[1][0] : &lAt[0][0];
    const u16* lB = (kt & 1) ? &lBt[1][0] : &lBt[0][0];

    const int kbase = 8 * hl;          // 16-bit A/B fragment K mapping (K=32 tile)
    v16bf af[2], bfv[4];
    #pragma unroll
    for (int mi = 0; mi < 2; ++mi)
      af[mi] = lds_frag(lA, wm * 32 + mi * 16 + lN, kbase);
    #pragma unroll
    for (int ni = 0; ni < 4; ++ni)
      bfv[ni] = lds_frag(lB, wn * 64 + ni * 16 + lN, kbase);
    #pragma unroll
    for (int mi = 0; mi < 2; ++mi)
      #pragma unroll
      for (int ni = 0; ni < 4; ++ni)
        acc[mi][ni] = __builtin_amdgcn_wmma_f32_16x16x32_bf16(
            false, af[mi], false, bfv[ni], (short)0, acc[mi][ni], false, false);
  }

  // ---- epilogue
  float* Cf = (float*)C + coff;
  u16*   Cb = (u16*)C + coff;
  #pragma unroll
  for (int mi = 0; mi < 2; ++mi) {
    #pragma unroll
    for (int ni = 0; ni < 4; ++ni) {
      AccU u; u.v = acc[mi][ni];
      const int colg  = n0 + wn * 64 + ni * 16 + lN;
      const int rbase = m0 + wm * 32 + mi * 16 + 8 * hl;
      #pragma unroll
      for (int e = 0; e < 8; ++e) {
        const int rowg = rbase + e;
        float val = u.f[e];
        if (flags & GF_RELU)   val = fmaxf(val, 0.f);
        if (flags & GF_CAUSAL) { if (colg >= rowg) val = 0.f; }
        if (flags & GF_AUX)    val *= bf2f(aux[xoff + (long long)rowg * ldaux + colg]);
        if (flags & GF_OUTBF16) Cb[(long long)rowg * ldc + colg] = f2bf(val);
        else                    Cf[(long long)rowg * ldc + colg] = val;
      }
    }
  }
}

// ---------------- transpose kernels (32x32 LDS tiles, coalesced both sides) ----
// out[c][r] = cvt(in[r][c]) per batch z
__global__ __launch_bounds__(256)
void transpose_cvt_f32(const float* __restrict__ in, u16* __restrict__ out,
                       int R, int C, long long sIn, long long sOut) {
  __shared__ u16 t[32][33];
  in  += (long long)blockIdx.z * sIn;
  out += (long long)blockIdx.z * sOut;
  const int c  = blockIdx.x * 32 + threadIdx.x;
  const int r0 = blockIdx.y * 32;
  #pragma unroll
  for (int i = 0; i < 4; ++i) {
    const int r = r0 + threadIdx.y + i * 8;
    t[threadIdx.y + i * 8][threadIdx.x] = f2bf(in[(long long)r * C + c]);
  }
  __syncthreads();
  const int oc = r0 + threadIdx.x;                 // output col = original row
  #pragma unroll
  for (int i = 0; i < 4; ++i) {
    const int orow = blockIdx.x * 32 + threadIdx.y + i * 8;   // = original col
    out[(long long)orow * R + oc] = t[threadIdx.x][threadIdx.y + i * 8];
  }
}

__global__ __launch_bounds__(256)
void transpose_bf16(const u16* __restrict__ in, u16* __restrict__ out,
                    int R, int C, long long sIn, long long sOut) {
  __shared__ u16 t[32][33];
  in  += (long long)blockIdx.z * sIn;
  out += (long long)blockIdx.z * sOut;
  const int c  = blockIdx.x * 32 + threadIdx.x;
  const int r0 = blockIdx.y * 32;
  #pragma unroll
  for (int i = 0; i < 4; ++i) {
    const int r = r0 + threadIdx.y + i * 8;
    t[threadIdx.y + i * 8][threadIdx.x] = in[(long long)r * C + c];
  }
  __syncthreads();
  const int oc = r0 + threadIdx.x;
  #pragma unroll
  for (int i = 0; i < 4; ++i) {
    const int orow = blockIdx.x * 32 + threadIdx.y + i * 8;
    out[(long long)orow * R + oc] = t[threadIdx.x][threadIdx.y + i * 8];
  }
}

// ---------------- LN / RoPE kernels ----------------
__global__ __launch_bounds__(256)
void embed_ln_kernel(const int* __restrict__ idx, const float* __restrict__ wte,
                     float* __restrict__ vf, u16* __restrict__ vb) {
  __shared__ float s1[256], s2[256];
  const int row = blockIdx.x;
  const int tid = threadIdx.x;
  const float* src = wte + (long long)idx[row] * Dm;
  float x[4], a = 0.f, b = 0.f;
  #pragma unroll
  for (int i = 0; i < 4; ++i) { x[i] = src[tid + i * 256]; a += x[i]; b += x[i] * x[i]; }
  s1[tid] = a; s2[tid] = b; __syncthreads();
  for (int o = 128; o > 0; o >>= 1) {
    if (tid < o) { s1[tid] += s1[tid + o]; s2[tid] += s2[tid + o]; }
    __syncthreads();
  }
  const float mean = s1[0] / Dm;
  const float rs   = rsqrtf(s2[0] / Dm - mean * mean + 1e-5f);
  #pragma unroll
  for (int i = 0; i < 4; ++i) {
    const float v = (x[i] - mean) * rs;
    vf[(long long)row * Dm + tid + i * 256] = v;
    vb[(long long)row * Dm + tid + i * 256] = f2bf(v);
  }
}

__global__ __launch_bounds__(256)
void ln_rows_kernel(const float* __restrict__ in, u16* __restrict__ out) {
  __shared__ float s1[256], s2[256];
  const int row = blockIdx.x;
  const int tid = threadIdx.x;
  const float* src = in + (long long)row * Dm;
  float x[4], a = 0.f, b = 0.f;
  #pragma unroll
  for (int i = 0; i < 4; ++i) { x[i] = src[tid + i * 256]; a += x[i]; b += x[i] * x[i]; }
  s1[tid] = a; s2[tid] = b; __syncthreads();
  for (int o = 128; o > 0; o >>= 1) {
    if (tid < o) { s1[tid] += s1[tid + o]; s2[tid] += s2[tid + o]; }
    __syncthreads();
  }
  const float mean = s1[0] / Dm;
  const float rs   = rsqrtf(s2[0] / Dm - mean * mean + 1e-5f);
  #pragma unroll
  for (int i = 0; i < 4; ++i)
    out[(long long)row * Dm + tid + i * 256] = f2bf((x[i] - mean) * rs);
}

__global__ __launch_bounds__(256)
void residual_ln_kernel(const float* __restrict__ ye, float* __restrict__ vf,
                        u16* __restrict__ vb) {
  __shared__ float s1[256], s2[256];
  const int row = blockIdx.x;
  const int tid = threadIdx.x;
  const float* y = ye + (long long)row * Dm;
  float* v = vf + (long long)row * Dm;
  float x[4], a = 0.f, b = 0.f;
  #pragma unroll
  for (int i = 0; i < 4; ++i) { x[i] = y[tid + i * 256]; a += x[i]; b += x[i] * x[i]; }
  s1[tid] = a; s2[tid] = b; __syncthreads();
  for (int o = 128; o > 0; o >>= 1) {
    if (tid < o) { s1[tid] += s1[tid + o]; s2[tid] += s2[tid + o]; }
    __syncthreads();
  }
  float mean = s1[0] / Dm;
  float rs   = rsqrtf(s2[0] / Dm - mean * mean + 1e-5f);
  float t[4]; a = 0.f; b = 0.f;
  __syncthreads();
  #pragma unroll
  for (int i = 0; i < 4; ++i) {
    t[i] = v[tid + i * 256] + (x[i] - mean) * rs;
    a += t[i]; b += t[i] * t[i];
  }
  s1[tid] = a; s2[tid] = b; __syncthreads();
  for (int o = 128; o > 0; o >>= 1) {
    if (tid < o) { s1[tid] += s1[tid + o]; s2[tid] += s2[tid + o]; }
    __syncthreads();
  }
  mean = s1[0] / Dm;
  rs   = rsqrtf(s2[0] / Dm - mean * mean + 1e-5f);
  #pragma unroll
  for (int i = 0; i < 4; ++i) {
    const float o = (t[i] - mean) * rs;
    v[tid + i * 256]  = o;
    vb[(long long)row * Dm + tid + i * 256] = f2bf(o);
  }
}

__global__ __launch_bounds__(256)
void rope_kernel(const u16* __restrict__ x, u16* __restrict__ xr) {
  const long long i = (long long)blockIdx.x * 256 + threadIdx.x;   // B*H*T*(SD/2)
  const int j  = (int)(i & (SDm / 2 - 1));
  const long long rowi = i >> 8;
  const int t  = (int)(rowi & (Tt - 1));
  const long long base = rowi * SDm;
  const float x1 = bf2f(x[base + 2 * j]);
  const float x2 = bf2f(x[base + 2 * j + 1]);
  const float invf = __expf(-__logf(10000.f) * (2.f * j) / (float)SDm);
  const float fr = (float)t * invf;
  const float c = __cosf(fr), s = __sinf(fr);
  xr[base + j]           = f2bf(x1 * c - x2 * s);
  xr[base + j + SDm / 2] = f2bf(x1 * s + x2 * c);
}

// ---------------- host launcher ----------------
extern "C" void kernel_launch(void* const* d_in, const int* in_sizes, int n_in,
                              void* d_out, int out_size, void* d_ws, size_t ws_size,
                              hipStream_t stream) {
  (void)in_sizes; (void)n_in; (void)out_size; (void)ws_size;
  const int*   idx      = (const int*)  d_in[0];
  const float* wte      = (const float*)d_in[1];
  const float* encoder  = (const float*)d_in[2];
  const float* dec_x    = (const float*)d_in[3];
  const float* dec_y    = (const float*)d_in[4];
  const float* readout  = (const float*)d_in[5];
  float* out = (float*)d_out;

  char* ws = (char*)d_ws;
  size_t off = 0;
  auto carve = [&](size_t bytes) -> char* {
    char* p = ws + off;
    off = (off + bytes + 255) & ~(size_t)255;
    return p;
  };
  float* vf    = (float*)carve((size_t)BT * Dm * 4);
  u16*   vb    = (u16*)  carve((size_t)BT * Dm * 2);
  u16*   vbT   = (u16*)  carve((size_t)BT * Dm * 2);          // per b: D x T
  u16*   xb    = (u16*)  carve((size_t)Bb * Hh * Tt * SDm * 2);
  u16*   xrb   = (u16*)  carve((size_t)Bb * Hh * Tt * SDm * 2);
  u16*   scb   = (u16*)  carve((size_t)Bb * Hh * Tt * Tt * 2);
  float* af    = (float*)carve((size_t)Bb * Hh * Tt * Dm * 4);
  u16*   alnb  = (u16*)  carve((size_t)Bb * Hh * Tt * Dm * 2);
  u16*   yb    = (u16*)  carve((size_t)BT * Nf * 2);
  float* yef   = (float*)carve((size_t)BT * Dm * 4);
  u16*   dxT   = (u16*)  carve((size_t)Hh * Dm * SDm * 2);    // per h: SD x D
  u16*   dyT   = (u16*)  carve((size_t)Hh * Dm * SDm * 2);    // per h: SD x D
  u16*   encT  = (u16*)  carve((size_t)Nf * Dm * 2);          // D x Nf
  u16*   roT   = (u16*)  carve((size_t)Dm * VOC * 2);         // VOC x D

  const dim3 blk(256);
  const dim3 tblk(32, 8);

  // weights: f32 -> bf16, transposed (so GEMM B-operand is always N x K)
  transpose_cvt_f32<<<dim3(SDm / 32, Dm / 32, Hh), tblk, 0, stream>>>(
      dec_x, dxT, Dm, SDm, (long long)Dm * SDm, (long long)SDm * Dm);
  transpose_cvt_f32<<<dim3(SDm / 32, Dm / 32, Hh), tblk, 0, stream>>>(
      dec_y, dyT, Dm, SDm, (long long)Dm * SDm, (long long)SDm * Dm);
  transpose_cvt_f32<<<dim3(Dm / 32, Nf / 32, 1), tblk, 0, stream>>>(
      encoder, encT, Nf, Dm, 0, 0);
  transpose_cvt_f32<<<dim3(VOC / 32, Dm / 32, 1), tblk, 0, stream>>>(
      readout, roT, Dm, VOC, 0, 0);

  // v = LN(wte[idx]); vbT = v^T per b
  embed_ln_kernel<<<BT, blk, 0, stream>>>(idx, wte, vf, vb);
  transpose_bf16<<<dim3(Dm / 32, Tt / 32, Bb), tblk, 0, stream>>>(
      vb, vbT, Tt, Dm, (long long)Tt * Dm, (long long)Dm * Tt);

  const long long TSD = (long long)Tt * SDm;
  const long long TD  = (long long)Tt * Dm;
  const long long TT  = (long long)Tt * Tt;

  for (int l = 0; l < Lit; ++l) {
    // x = relu(v @ dec_x[h])  -> bf16 (B,H,T,SD)
    gemm_bf16_wmma<<<dim3(SDm / BN, Tt / BM, Bb * Hh), blk, 0, stream>>>(
        vb, dxT, xb, nullptr, Tt, SDm, Dm, Dm, Dm, SDm, 0, Hh,
        TD, 0,  0, (long long)SDm * Dm,  (long long)Hh * TSD, TSD,  0, 0,
        GF_RELU | GF_OUTBF16);

    // xr = rope(x)
    rope_kernel<<<(Bb * Hh * Tt * (SDm / 2)) / 256, blk, 0, stream>>>(xb, xrb);

    // scores = tril(xr @ xr^T, k=-1)  -> bf16 (B,H,T,T); B-operand already N x K
    gemm_bf16_wmma<<<dim3(Tt / BN, Tt / BM, Bb * Hh), blk, 0, stream>>>(
        xrb, xrb, scb, nullptr, Tt, Tt, SDm, SDm, SDm, Tt, 0, Hh,
        (long long)Hh * TSD, TSD,  (long long)Hh * TSD, TSD,
        (long long)Hh * TT,  TT,   0, 0,
        GF_CAUSAL | GF_OUTBF16);

    // a = scores @ v  -> f32 (B,H,T,D); B-operand = vbT (D x T) per b
    gemm_bf16_wmma<<<dim3(Dm / BN, Tt / BM, Bb * Hh), blk, 0, stream>>>(
        scb, vbT, af, nullptr, Tt, Dm, Tt, Tt, Tt, Dm, 0, Hh,
        (long long)Hh * TT, TT,  (long long)Dm * Tt, 0,
        (long long)Hh * TD, TD,  0, 0,
        0);

    // a_ln = LN(a) -> bf16
    ln_rows_kernel<<<Bb * Hh * Tt, blk, 0, stream>>>(af, alnb);

    // y = relu(a_ln @ dec_y[h]) * x  -> bf16 stored as (B,T,H*SD) = (BT, N)
    gemm_bf16_wmma<<<dim3(SDm / BN, Tt / BM, Bb * Hh), blk, 0, stream>>>(
        alnb, dyT, yb, xb, Tt, SDm, Dm, Dm, Dm, Nf, SDm, Hh,
        (long long)Hh * TD, TD,  0, (long long)SDm * Dm,
        (long long)Tt * Nf, SDm,  (long long)Hh * TSD, TSD,
        GF_RELU | GF_OUTBF16 | GF_AUX);

    // ye = y @ encoder  -> f32 (BT, D); B-operand = encT (D x Nf)
    gemm_bf16_wmma<<<dim3(Dm / BN, BT / BM, 1), blk, 0, stream>>>(
        yb, encT, yef, nullptr, BT, Dm, Nf, Nf, Nf, Dm, 0, 1,
        0, 0, 0, 0, 0, 0, 0, 0, 0);

    // v = LN(v + LN(ye)); refresh vbT
    residual_ln_kernel<<<BT, blk, 0, stream>>>(yef, vf, vb);
    transpose_bf16<<<dim3(Dm / 32, Tt / 32, Bb), tblk, 0, stream>>>(
        vb, vbT, Tt, Dm, (long long)Tt * Dm, (long long)Dm * Tt);
  }

  // logits = v @ readout  -> f32 (BT, VOCAB); B-operand = roT (VOC x D)
  gemm_bf16_wmma<<<dim3(VOC / BN, BT / BM, 1), blk, 0, stream>>>(
      vb, roT, out, nullptr, BT, VOC, Dm, Dm, Dm, VOC, 0, 1,
      0, 0, 0, 0, 0, 0, 0, 0, 0);
}
